// DH_LIFNode_86474871538276
// MI455X (gfx1250) — matrix operand order repaired and checked by reading, not compiled, not run on hardware
//
#include <hip/hip_runtime.h>
#include <hip/hip_bf16.h>

#define BATCH  512
#define INPUT  4096
#define HIDDEN 2048
#define NB     4

typedef __attribute__((ext_vector_type(16))) _Float16 v16h;
typedef __attribute__((ext_vector_type(8)))  _Float16 v8h;
typedef __attribute__((ext_vector_type(4)))  _Float16 h4;
typedef __attribute__((ext_vector_type(8)))  float    v8f;

// ---------------------------------------------------------------------------
// Kernel 1: fold mask + tau_n into an effective weight matrix, cast to f16.
//   Weff[h,i] = W[h,i] + 0.5 * sum_k (1 - sigmoid(tau_n[h,k])) * mask[h,i,k]
// Reads mask (128 MB) exactly once -> this is the HBM-bound pass.
// ---------------------------------------------------------------------------
__global__ __launch_bounds__(256) void fuse_weights(
    const float* __restrict__ W,
    const float* __restrict__ tau_n,
    const float* __restrict__ mask,
    _Float16* __restrict__ Weff) {
  const int idx = blockIdx.x * 256 + threadIdx.x;   // idx = h*INPUT + i
  const int h   = idx >> 12;                        // INPUT = 4096
  const float4 m4 = reinterpret_cast<const float4*>(mask)[idx]; // mask[h,i,0..3]
  const float  w  = W[idx];
  float acc = 0.0f;
#pragma unroll
  for (int k = 0; k < NB; ++k) {
    const float a  = 1.0f / (1.0f + __expf(-tau_n[h * NB + k])); // sigmoid
    const float mk = (&m4.x)[k];
    acc += (1.0f - a) * mk;
  }
  Weff[idx] = (_Float16)(w + 0.5f * acc);
}

// ---------------------------------------------------------------------------
// Kernel 2: cast x (exact 0/1 values) to f16, 4 elements per thread.
// ---------------------------------------------------------------------------
__global__ __launch_bounds__(256) void cvt_x(
    const float* __restrict__ x, _Float16* __restrict__ x16) {
  const int idx = blockIdx.x * 256 + threadIdx.x;    // over BATCH*INPUT/4
  const float4 v = reinterpret_cast<const float4*>(x)[idx];
  h4 o;
  o.x = (_Float16)v.x; o.y = (_Float16)v.y;
  o.z = (_Float16)v.z; o.w = (_Float16)v.w;
  reinterpret_cast<h4*>(x16)[idx] = o;
}

// ---------------------------------------------------------------------------
// Fragment loaders; A and B share one contiguous K order.
// A (16x32 f16): lane holds row M=lane&15; K 0..15 (lanes 0-15)
//                or K 16..31 (lanes 16-31) -> two b128 loads.
// B (32x16 f16): lane holds col N=lane&15; 16 contiguous K values
//                (K+0..15 / K+16..31) -> one 32B contiguous load.
// ---------------------------------------------------------------------------
__device__ __forceinline__ v16h load_a_frag(const _Float16* p, int k, int akoff) {
  const v8h lo = *reinterpret_cast<const v8h*>(p + k + akoff);
  const v8h hi = *reinterpret_cast<const v8h*>(p + k + 8 + akoff);
  return __builtin_shufflevector(lo, hi,
      0, 1, 2, 3, 4, 5, 6, 7, 8, 9, 10, 11, 12, 13, 14, 15);
}

__device__ __forceinline__ v16h load_b_frag(const _Float16* p, int k, int bkoff) {
  return *reinterpret_cast<const v16h*>(p + k + bkoff);
}

// ---------------------------------------------------------------------------
// Kernel 3: WMMA GEMM  C[512,2048] = x16 @ Weff^T  with fused LIF epilogue.
// Wave tile: 32(M) x 32(N): acc[2][2] (32 VGPRs) + ping-pong fragments
// (64 VGPRs) + 4 incrementing pointers -> ~130 VGPRs, so the allocator keeps
// accumulators and fragments pinned (no spill/juggle copies) and ~7 waves/SIMD
// remain for latency hiding. K loop unrolled x2 with ping-pong buffers.
// Block: 256 threads = 8 waves arranged 2(M) x 4(N) -> 64 x 128 block tile.
// Grid: (2048/128, 512/64) = (16, 8).
// ---------------------------------------------------------------------------
__global__ __launch_bounds__(256) void gemm_spike(
    const _Float16* __restrict__ A,      // x16   [512 , 4096] row-major
    const _Float16* __restrict__ Bm,     // Weff  [2048, 4096] row-major (B = Weff^T)
    const float*    __restrict__ bias,   // b     [2048]
    const float*    __restrict__ tau_m,  // tau_m [2048]
    float*          __restrict__ out) {  // spike [512 , 2048]
  const int lane = threadIdx.x & 31;
  const int wave = threadIdx.x >> 5;
  const int wm   = wave >> 2;            // 0..1
  const int wn   = wave & 3;             // 0..3

  const int mbase = blockIdx.y * 64  + wm * 32;   // wave rows mbase..mbase+31
  const int nbase = blockIdx.x * 128 + wn * 32;   // wave cols nbase..nbase+31

  const int hi16  = (lane >= 16) ? 1 : 0;
  const int akoff = hi16 * 16;           // A: upper half-wave holds K+16..31
  const int bkoff = hi16 * 16;           // B: upper half-wave holds K+16..31

  const _Float16* ap0 = A + (size_t)(mbase + 0  + (lane & 15)) * INPUT;
  const _Float16* ap1 = A + (size_t)(mbase + 16 + (lane & 15)) * INPUT;
  const _Float16* bp0 = Bm + (size_t)(nbase +  0 + (lane & 15)) * INPUT;
  const _Float16* bp1 = Bm + (size_t)(nbase + 16 + (lane & 15)) * INPUT;

  v8f acc[2][2];
#pragma unroll
  for (int mi = 0; mi < 2; ++mi)
#pragma unroll
    for (int j = 0; j < 2; ++j) acc[mi][j] = (v8f)0.0f;

  // ---- software pipeline: ping-pong buffers, incrementing pointers ----
  v16h aA[2], bA[2], aB[2], bB[2];
  aA[0] = load_a_frag(ap0, 0, akoff);
  aA[1] = load_a_frag(ap1, 0, akoff);
  bA[0] = load_b_frag(bp0, 0, bkoff);
  bA[1] = load_b_frag(bp1, 0, bkoff);

  // 63 iterations x 64 K + peeled tail (2 steps) = 128 K-steps of 32.
#pragma unroll 1
  for (int it = 0; it < (INPUT / 64) - 1; ++it) {
    // streaming prefetch for the first (HBM) pass of the B operand
    __builtin_prefetch(bp0 + 1024, 0, 1);
    __builtin_prefetch(bp1 + 1024, 0, 1);

    // fetch odd step (k+32) into B-buffers
    aB[0] = load_a_frag(ap0, 32, akoff);
    aB[1] = load_a_frag(ap1, 32, akoff);
    bB[0] = load_b_frag(bp0, 32, bkoff);
    bB[1] = load_b_frag(bp1, 32, bkoff);

    // multiply even step (k)
#pragma unroll
    for (int mi = 0; mi < 2; ++mi)
#pragma unroll
      for (int j = 0; j < 2; ++j)
        acc[mi][j] = __builtin_amdgcn_wmma_f32_16x16x32_f16(
            false, aA[mi], false, bA[j], (short)0, acc[mi][j], false, false);

    // fetch next even step (k+64) into A-buffers
    aA[0] = load_a_frag(ap0, 64, akoff);
    aA[1] = load_a_frag(ap1, 64, akoff);
    bA[0] = load_b_frag(bp0, 64, bkoff);
    bA[1] = load_b_frag(bp1, 64, bkoff);

    // multiply odd step (k+32)
#pragma unroll
    for (int mi = 0; mi < 2; ++mi)
#pragma unroll
      for (int j = 0; j < 2; ++j)
        acc[mi][j] = __builtin_amdgcn_wmma_f32_16x16x32_f16(
            false, aB[mi], false, bB[j], (short)0, acc[mi][j], false, false);

    ap0 += 64; ap1 += 64; bp0 += 64; bp1 += 64;
  }

  // ---- peeled tail: two K-steps remain, A-buffers hold the first ----
  aB[0] = load_a_frag(ap0, 32, akoff);
  aB[1] = load_a_frag(ap1, 32, akoff);
  bB[0] = load_b_frag(bp0, 32, bkoff);
  bB[1] = load_b_frag(bp1, 32, bkoff);
#pragma unroll
  for (int mi = 0; mi < 2; ++mi)
#pragma unroll
    for (int j = 0; j < 2; ++j)
      acc[mi][j] = __builtin_amdgcn_wmma_f32_16x16x32_f16(
          false, aA[mi], false, bA[j], (short)0, acc[mi][j], false, false);
#pragma unroll
  for (int mi = 0; mi < 2; ++mi)
#pragma unroll
    for (int j = 0; j < 2; ++j)
      acc[mi][j] = __builtin_amdgcn_wmma_f32_16x16x32_f16(
          false, aB[mi], false, bB[j], (short)0, acc[mi][j], false, false);

  // Epilogue: v = (1 - sigmoid(tau_m[n])) * (acc + b[n]);  spike = (v >= 1).
  // C/D layout: VGPR r -> row tile_m + r + (lane>=16 ? 8 : 0), col = lane&15.
  const int ncol = lane & 15;
#pragma unroll
  for (int j = 0; j < 2; ++j) {
    const int   n     = nbase + j * 16 + ncol;
    const float beta  = 1.0f / (1.0f + __expf(-tau_m[n]));
    const float scale = 1.0f - beta;
    const float bb    = bias[n];
#pragma unroll
    for (int mi = 0; mi < 2; ++mi) {
      const int rbase = mbase + mi * 16 + (hi16 ? 8 : 0);
#pragma unroll
      for (int r = 0; r < 8; ++r) {
        const float v = scale * (acc[mi][j][r] + bb);
        out[(size_t)(rbase + r) * HIDDEN + n] = (v >= 1.0f) ? 1.0f : 0.0f;
      }
    }
  }
}

// ---------------------------------------------------------------------------
// Host-side launcher. Inputs (setup_inputs order): x, W, b, tau_m, tau_n, mask.
// Workspace: [Weff f16: 16 MB][x16 f16: 4 MB].
// ---------------------------------------------------------------------------
extern "C" void kernel_launch(void* const* d_in, const int* in_sizes, int n_in,
                              void* d_out, int out_size, void* d_ws, size_t ws_size,
                              hipStream_t stream) {
  const float* x     = (const float*)d_in[0];
  const float* W     = (const float*)d_in[1];
  const float* b     = (const float*)d_in[2];
  const float* tau_m = (const float*)d_in[3];
  const float* tau_n = (const float*)d_in[4];
  const float* mask  = (const float*)d_in[5];
  float* out = (float*)d_out;

  _Float16* Weff = (_Float16*)d_ws;
  _Float16* x16  = (_Float16*)((char*)d_ws + (size_t)HIDDEN * INPUT * sizeof(_Float16));

  // 1) Fold mask/tau_n/W into f16 effective weights (single streaming pass).
  fuse_weights<<<(HIDDEN * INPUT) / 256, 256, 0, stream>>>(W, tau_n, mask, Weff);

  // 2) Cast x to f16.
  cvt_x<<<(BATCH * INPUT / 4) / 256, 256, 0, stream>>>(x, x16);

  // 3) WMMA GEMM + fused spike epilogue.
  dim3 grid(HIDDEN / 128, BATCH / 64);
  gemm_spike<<<grid, 256, 0, stream>>>(x16, Weff, b, tau_m, out);
}
